// DiffusionNetBlockSNK_75917841924408
// MI455X (gfx1250) — matrix operand order repaired and checked by reading, not compiled, not run on hardware
//
#include <hip/hip_runtime.h>
#include <math.h>

// ---------------------------------------------------------------------------
// CDNA5 (gfx1250) DiffusionNet block.
// One strided-batched bf16 WMMA GEMM (v_wmma_f32_16x16x32_bf16, wave32):
//   256 threads = 8 waves, 128x128 block tile, K-step 32,
//   double-buffered LDS staging (f32 -> bf16 on the fly),
//   fragment preloading so the 8 WMMAs issue back-to-back per K-step.
// ---------------------------------------------------------------------------

typedef __attribute__((ext_vector_type(16))) __bf16 v16bf;
typedef __attribute__((ext_vector_type(8)))  __bf16 v8bf;
typedef __attribute__((ext_vector_type(8)))  float  v8f;

#define GEMM_TM 128           // M rows per block (8 waves x 16)
#define GEMM_TN 128           // N cols per block (8 WMMA tiles per wave)
#define GEMM_TK 32            // K per step

// ---------------------------------------------------------------------------
// C[b] (+)= (A[b] @ bscale*B[b]) (+bias)
//   A element (i,k) at A[b*bsA + i*sAr + k*sAc]
//   B element (k,j) at B[b*bsB + k*sBr + j*sBc]
//   C element (i,j) at C[b*bsC + i*ldC + j]
// Requires: M % 128 == 0, N % 128 == 0, Kd % 32 == 0  (true for all calls).
// ---------------------------------------------------------------------------
__global__ void __launch_bounds__(256)
gemm_bf16_wmma(const float* __restrict__ A, int sAr, int sAc, int bsA,
               const float* __restrict__ Bm, int sBr, int sBc, int bsB,
               float* Cm, int ldC, int bsC,
               const float* __restrict__ bias, int Kd, float bscale, int accum)
{
    __shared__ __attribute__((aligned(32))) __bf16 sA[2][GEMM_TM * GEMM_TK]; // [m][k]
    __shared__ __attribute__((aligned(32))) __bf16 sB[2][GEMM_TN * GEMM_TK]; // [n][k]

    const int tid  = threadIdx.x;
    const int lane = tid & 31;
    const int wave = tid >> 5;
    const int row  = lane & 15;   // m (A) / n (B,C) within 16x16 tile
    const int hi   = lane >> 4;   // lane half

    const int bA = blockIdx.z * bsA;
    const int bB = blockIdx.z * bsB;
    const int bC = blockIdx.z * bsC;
    const int nBase = blockIdx.x * GEMM_TN;
    const int m0    = blockIdx.y * GEMM_TM;

    // ---- accumulator init: 8 tiles of 16x16 per wave ----
    v8f acc[8];
    const int mTile = m0 + wave * 16;
    if (accum) {
        #pragma unroll
        for (int j = 0; j < 8; ++j)
            #pragma unroll
            for (int r = 0; r < 8; ++r)
                acc[j][r] = Cm[bC + (mTile + hi * 8 + r) * ldC + (nBase + 16 * j + row)];
    } else if (bias) {
        #pragma unroll
        for (int j = 0; j < 8; ++j) {
            float bv = bias[nBase + 16 * j + row];
            #pragma unroll
            for (int r = 0; r < 8; ++r) acc[j][r] = bv;
        }
    } else {
        #pragma unroll
        for (int j = 0; j < 8; ++j)
            #pragma unroll
            for (int r = 0; r < 8; ++r) acc[j][r] = 0.0f;
    }

    // ---- cooperative staging (coalesced for either memory order) ----
    auto stageA = [&](int buf, int k0) {
        __bf16* dst = &sA[buf][0];
        if (sAc == 1) {               // A row-major: 32-wide K runs
            int k = tid & 31, m = tid >> 5;                 // 8 rows/pass
            const float* src = A + bA + (m0 + m) * sAr + (k0 + k);
            #pragma unroll
            for (int it = 0; it < 16; ++it)
                dst[(m + it * 8) * GEMM_TK + k] = (__bf16)src[it * 8 * sAr];
        } else {                      // A i-contiguous (transposed use)
            int m = tid & 127, k = tid >> 7;                // 2 k/pass
            const float* src = A + bA + (m0 + m) * sAr + (k0 + k) * sAc;
            #pragma unroll
            for (int it = 0; it < 16; ++it)
                dst[m * GEMM_TK + (k + it * 2)] = (__bf16)src[it * 2 * sAc];
        }
    };
    auto stageB = [&](int buf, int k0) {
        __bf16* dst = &sB[buf][0];
        if (sBc == 1) {               // B row-major: n-contiguous
            int n = tid & 127, k = tid >> 7;                // 2 k/pass
            const float* src = Bm + bB + (k0 + k) * sBr + (nBase + n);
            #pragma unroll
            for (int it = 0; it < 16; ++it)
                dst[n * GEMM_TK + (k + it * 2)] = (__bf16)(bscale * src[it * 2 * sBr]);
        } else {                      // B k-contiguous (W^T use)
            int k = tid & 31, n = tid >> 5;                 // 8 n/pass
            const float* src = Bm + bB + (k0 + k) * sBr + (nBase + n) * sBc;
            #pragma unroll
            for (int it = 0; it < 16; ++it)
                dst[(n + it * 8) * GEMM_TK + k] = (__bf16)(bscale * src[it * 8 * sBc]);
        }
    };

    const int aoff = (wave * 16 + row) * GEMM_TK + hi * 8;

    int cur = 0;
    stageA(0, 0);
    stageB(0, 0);
    __syncthreads();

    for (int k0 = 0; k0 < Kd; k0 += GEMM_TK) {
        int nxt = cur ^ 1;

        // 1) pull ALL fragments for this step into VGPRs first, so one
        //    in-order DScnt wait covers them and the WMMAs run back-to-back.
        v8bf alo = *(const v8bf*)(&sA[cur][0] + aoff);
        v8bf ahi = *(const v8bf*)(&sA[cur][0] + aoff + 16);
        v16bf bfr[8];
        #pragma unroll
        for (int j = 0; j < 8; ++j)
            bfr[j] = __builtin_shufflevector(
                *(const v8bf*)(&sB[cur][0] + (16 * j + row) * GEMM_TK + hi * 8),
                *(const v8bf*)(&sB[cur][0] + (16 * j + row) * GEMM_TK + hi * 8 + 16),
                0, 1, 2, 3, 4, 5, 6, 7, 8, 9, 10, 11, 12, 13, 14, 15);
        __builtin_amdgcn_sched_barrier(0);

        // 2) stage next K tile: global loads overlap the WMMA burst below,
        //    its ds_stores trail behind the fragment ds_loads in DScnt order.
        if (k0 + GEMM_TK < Kd) {
            stageA(nxt, k0 + GEMM_TK);
            stageB(nxt, k0 + GEMM_TK);
        }

        // 3) 8 WMMAs with a single A fragment (ISA 16x32 bf16 layout).
        v16bf afrag = __builtin_shufflevector(alo, ahi,
            0, 1, 2, 3, 4, 5, 6, 7, 8, 9, 10, 11, 12, 13, 14, 15);
        #pragma unroll
        for (int j = 0; j < 8; ++j)
            acc[j] = __builtin_amdgcn_wmma_f32_16x16x32_bf16(
                false, afrag, false, bfr[j], (short)0, acc[j], false, false);

        __syncthreads();
        cur = nxt;
    }

    #pragma unroll
    for (int j = 0; j < 8; ++j)
        #pragma unroll
        for (int r = 0; r < 8; ++r)
            Cm[bC + (mTile + hi * 8 + r) * ldC + (nBase + 16 * j + row)] = acc[j][r];
}

// ---------------------------------------------------------------------------
// Elementwise / reduction / sparse kernels
// ---------------------------------------------------------------------------
__global__ void fill0_kernel(float* p, long n) {
    long i = (long)blockIdx.x * blockDim.x + threadIdx.x;
    if (i < n) p[i] = 0.0f;
}

__global__ void ew_mass_mul(const float* __restrict__ x, const float* __restrict__ mass,
                            float* __restrict__ outp, long total, int C) {
    long i = (long)blockIdx.x * blockDim.x + threadIdx.x;
    if (i < total) outp[i] = x[i] * mass[i / C];
}

__global__ void ew_coef(float* xs, const float* __restrict__ evals,
                        const float* __restrict__ tptr, long total, int C) {
    long i = (long)blockIdx.x * blockDim.x + threadIdx.x;
    if (i < total) {
        long bk = i / C;
        int  c  = (int)(i % C);
        float t = fmaxf(tptr[c], 1e-8f);
        xs[i] *= expf(-evals[bk] * t);
    }
}

// COO SpMM: out[row,:] += val * dense[col,:]   (one block per nonzero)
__global__ void spmm_coo(const int* __restrict__ rows, const int* __restrict__ cols,
                         const float* __restrict__ vals, const float* __restrict__ dense,
                         float* outp, int C) {
    int e = blockIdx.x;
    int c = threadIdx.x;
    int r  = rows[e];
    int cl = cols[e];
    float v = vals[e];
    atomicAdd(&outp[(long)r * C + c], v * dense[(long)cl * C + c]);
}

__global__ void ew_gradfeat(const float* gX, const float* gY,
                            const float* bre, const float* bim,
                            float* outp, long total) {
    long i = (long)blockIdx.x * blockDim.x + threadIdx.x;
    if (i < total) outp[i] = tanhf(gX[i] * bre[i] + gY[i] * bim[i]);
}

// per-channel sum / sumsq partial reduction (blockDim.x == C)
__global__ void bn_stats(const float* __restrict__ h, float* stats,
                         long Nrows, int rowsPerBlock) {
    int c = threadIdx.x;
    int C = blockDim.x;
    long r0 = (long)blockIdx.x * rowsPerBlock;
    float s = 0.0f, ss = 0.0f;
    for (int r = 0; r < rowsPerBlock; ++r) {
        float v = h[(r0 + r) * C + c];
        s += v;
        ss += v * v;
    }
    atomicAdd(&stats[c], s);
    atomicAdd(&stats[C + c], ss);
}

__global__ void bn_apply(const float* h, const float* __restrict__ stats,
                         const float* __restrict__ g, const float* __restrict__ be,
                         const float* __restrict__ resid, float* outp,
                         long Nrows, int C, int relu) {
    long total = Nrows * (long)C;
    long i = (long)blockIdx.x * blockDim.x + threadIdx.x;
    if (i >= total) return;
    int c = (int)(i % C);
    float inv = 1.0f / (float)Nrows;
    float m   = stats[c] * inv;
    float var = stats[C + c] * inv - m * m;
    float v = (h[i] - m) * rsqrtf(var + 1e-5f) * g[c] + be[c];
    if (relu) v = fmaxf(v, 0.0f);
    if (resid) v += resid[i];
    outp[i] = v;
}

// ---------------------------------------------------------------------------
// Host orchestration
// ---------------------------------------------------------------------------
extern "C" void kernel_launch(void* const* d_in, const int* in_sizes, int n_in,
                              void* d_out, int out_size, void* d_ws, size_t ws_size,
                              hipStream_t stream) {
    (void)n_in; (void)out_size; (void)ws_size;

    const float* x     = (const float*)d_in[0];
    const float* mass  = (const float*)d_in[1];
    const float* evals = (const float*)d_in[2];
    const float* evecs = (const float*)d_in[3];
    const int*   gxr   = (const int*)  d_in[4];
    const int*   gxc   = (const int*)  d_in[5];
    const float* gxv   = (const float*)d_in[6];
    const int*   gyr   = (const int*)  d_in[7];
    const int*   gyc   = (const int*)  d_in[8];
    const float* gyv   = (const float*)d_in[9];
    const float* tdiff = (const float*)d_in[10];
    const float* A_re  = (const float*)d_in[11];
    const float* A_im  = (const float*)d_in[12];
    const float* W0 = (const float*)d_in[13]; const float* b0 = (const float*)d_in[14];
    const float* g0 = (const float*)d_in[15]; const float* be0 = (const float*)d_in[16];
    const float* W1 = (const float*)d_in[17]; const float* b1 = (const float*)d_in[18];
    const float* g1 = (const float*)d_in[19]; const float* be1 = (const float*)d_in[20];
    const float* W2 = (const float*)d_in[21]; const float* b2 = (const float*)d_in[22];
    const float* g2 = (const float*)d_in[23]; const float* be2 = (const float*)d_in[24];
    float* out = (float*)d_out;

    // Derived shapes (host-side, from in_sizes): C=256, N=65536, K=128, B=8
    const int  C   = in_sizes[10];
    const long N   = in_sizes[1];
    const int  K   = (int)((long)in_sizes[3] / N);
    const int  Bb  = in_sizes[2] / K;
    const long V   = N / Bb;
    const long NNZ = in_sizes[4];
    const long NC  = N * (long)C;
    const long BKC = (long)Bb * K * C;

    // Workspace layout (reuse across phases)
    float* ws   = (float*)d_ws;
    float* bufA = ws;             // mx   -> gX
    float* bufB = ws + NC;        // gY   -> h0
    float* bufC = ws + 2 * NC;    // bre  -> grad_feat
    float* bufD = ws + 3 * NC;    // bim  -> h1
    float* bufE = ws + 4 * NC;    // x_diffuse -> h2
    float* xs   = ws + 5 * NC;    // (B*K, C) spectral coeffs
    float* stats = xs + BKC;      // 2*C

    const int TPB = 256;
    auto blocks = [&](long n) { return (unsigned)((n + TPB - 1) / TPB); };

    auto gemm = [&](const float* Am, int sAr, int sAc, int bsA,
                    const float* Bm, int sBr, int sBc, int bsB,
                    float* Cm, int ldC, int bsC,
                    const float* bias, long M, int Nn, int Kd, int nb,
                    float bscale, int accum) {
        dim3 grid((unsigned)(Nn / GEMM_TN), (unsigned)(M / GEMM_TM), (unsigned)nb);
        gemm_bf16_wmma<<<grid, dim3(256), 0, stream>>>(
            Am, sAr, sAc, bsA, Bm, sBr, sBc, bsB, Cm, ldC, bsC, bias, Kd, bscale, accum);
    };

    // 1) mx = mass[:,None] * x
    ew_mass_mul<<<blocks(NC), TPB, 0, stream>>>(x, mass, bufA, NC, C);

    // 2) x_spec[b] = evecs[b]^T @ mx[b]   (K x V) @ (V x C), batched
    gemm(evecs, 1, K, (int)(V * K),
         bufA, C, 1, (int)(V * C),
         xs, C, K * C,
         nullptr, K, C, (int)V, Bb, 1.0f, 0);

    // 3) xs *= exp(-evals * t)
    ew_coef<<<blocks(BKC), TPB, 0, stream>>>(xs, evals, tdiff, BKC, C);

    // 4) x_diffuse[b] = evecs[b] @ xs[b]  (V x K) @ (K x C), batched
    gemm(evecs, K, 1, (int)(V * K),
         xs, C, 1, K * C,
         bufE, C, (int)(V * C),
         nullptr, V, C, K, Bb, 1.0f, 0);

    // 5) gX = spmm(gx, x_diffuse); gY = spmm(gy, x_diffuse)
    fill0_kernel<<<blocks(2 * NC), TPB, 0, stream>>>(bufA, 2 * NC);
    spmm_coo<<<(unsigned)NNZ, C, 0, stream>>>(gxr, gxc, gxv, bufE, bufA, C);
    spmm_coo<<<(unsigned)NNZ, C, 0, stream>>>(gyr, gyc, gyv, bufE, bufB, C);

    // 6) bre = gX @ A_re^T - gY @ A_im^T ; bim = gY @ A_re^T + gX @ A_im^T
    gemm(bufA, C, 1, 0, A_re, 1, C, 0, bufC, C, 0, nullptr, N, C, C, 1,  1.0f, 0);
    gemm(bufB, C, 1, 0, A_im, 1, C, 0, bufC, C, 0, nullptr, N, C, C, 1, -1.0f, 1);
    gemm(bufB, C, 1, 0, A_re, 1, C, 0, bufD, C, 0, nullptr, N, C, C, 1,  1.0f, 0);
    gemm(bufA, C, 1, 0, A_im, 1, C, 0, bufD, C, 0, nullptr, N, C, C, 1,  1.0f, 1);

    // 7) grad_feat = tanh(gX*bre + gY*bim)   (overwrites bre slot)
    ew_gradfeat<<<blocks(NC), TPB, 0, stream>>>(bufA, bufB, bufC, bufD, bufC, NC);

    // 8) h0 = x @ W0[0:C] + b0 + x_diffuse @ W0[C:2C] + grad_feat @ W0[2C:3C]
    gemm(x,    C, 1, 0, W0,             C, 1, 0, bufB, C, 0, b0,      N, C, C, 1, 1.0f, 0);
    gemm(bufE, C, 1, 0, W0 + C * C,     C, 1, 0, bufB, C, 0, nullptr, N, C, C, 1, 1.0f, 1);
    gemm(bufC, C, 1, 0, W0 + 2 * C * C, C, 1, 0, bufB, C, 0, nullptr, N, C, C, 1, 1.0f, 1);

    // 9) BN0 + ReLU (in place)
    fill0_kernel<<<blocks(2 * C), TPB, 0, stream>>>(stats, 2 * C);
    bn_stats<<<(unsigned)(N / 256), C, 0, stream>>>(bufB, stats, N, 256);
    bn_apply<<<blocks(NC), TPB, 0, stream>>>(bufB, stats, g0, be0, nullptr, bufB, N, C, 1);

    // 10) h1 = h0 @ W1 + b1 ; BN1 + ReLU
    gemm(bufB, C, 1, 0, W1, C, 1, 0, bufD, C, 0, b1, N, C, C, 1, 1.0f, 0);
    fill0_kernel<<<blocks(2 * C), TPB, 0, stream>>>(stats, 2 * C);
    bn_stats<<<(unsigned)(N / 256), C, 0, stream>>>(bufD, stats, N, 256);
    bn_apply<<<blocks(NC), TPB, 0, stream>>>(bufD, stats, g1, be1, nullptr, bufD, N, C, 1);

    // 11) h2 = h1 @ W2 + b2 ; BN2 (no relu) + residual x -> out
    gemm(bufD, C, 1, 0, W2, C, 1, 0, bufE, C, 0, b2, N, C, C, 1, 1.0f, 0);
    fill0_kernel<<<blocks(2 * C), TPB, 0, stream>>>(stats, 2 * C);
    bn_stats<<<(unsigned)(N / 256), C, 0, stream>>>(bufE, stats, N, 256);
    bn_apply<<<blocks(NC), TPB, 0, stream>>>(bufE, stats, g2, be2, x, out, N, C, 0);
}
